// SinkhornLoss_6090263625749
// MI455X (gfx1250) — matrix operand order repaired and checked by reading, not compiled, not run on hardware
//
#include <hip/hip_runtime.h>

#define N 8192
#define DIMS 32
#define EPS 0.1f
#define STABF 1e-8f
#define NUM_ITERS 100

typedef __attribute__((ext_vector_type(16))) _Float16 v16h;
typedef __attribute__((ext_vector_type(8)))  float    v8f;
typedef __attribute__((ext_vector_type(8)))  _Float16 h8;
typedef __attribute__((ext_vector_type(4)))  float    f4;

// ---------------------------------------------------------------------------
// Row squared-norms: one thread per row (32 floats = 2 cachelines per row).
__global__ void norms_kernel(const float* __restrict__ x, float* __restrict__ out) {
    int i = blockIdx.x * blockDim.x + threadIdx.x;
    if (i < N) {
        const f4* p = (const f4*)(x + (size_t)i * DIMS);
        float s = 0.f;
#pragma unroll
        for (int t = 0; t < 8; ++t) {
            f4 v = p[t];
            s += v.x * v.x + v.y * v.y + v.z * v.z + v.w * v.w;
        }
        out[i] = s;
    }
}

// ---------------------------------------------------------------------------
// K[i][j] = exp(-max(na[i]+nb[j]-2*xa_i.xb_j, 0)/EPS), stored fp16.
// One wave per 16x16 tile; single v_wmma_f32_16x16x32_f16 covers dim=32.
__global__ void gram_exp_kernel(const float* __restrict__ xa, const float* __restrict__ xb,
                                const float* __restrict__ na, const float* __restrict__ nb,
                                _Float16* __restrict__ Kout) {
    int wave = (blockIdx.x * blockDim.x + threadIdx.x) >> 5;   // 0 .. 512*512-1
    int lane = threadIdx.x & 31;
    int ti = wave >> 9;          // row tile
    int tj = wave & 511;         // col tile
    int m  = lane & 15;
    int h  = lane >> 4;

    const float* pa = xa + (size_t)(ti * 16 + m) * DIMS;
    const float* pb = xb + (size_t)(tj * 16 + m) * DIMS;

    // A fragment (16x32 f16): lane(m,h) holds K = {8h..8h+7} and {16+8h..16+8h+7}
    v16h a, b;
#pragma unroll
    for (int t = 0; t < 8; ++t)  a[t]     = (_Float16)pa[8 * h + t];
#pragma unroll
    for (int t = 0; t < 8; ++t)  a[8 + t] = (_Float16)pa[16 + 8 * h + t];
    // B fragment (32x16 f16): lane(n=m,h) holds K = {8h..8h+7} and {16+8h..16+8h+7}, B[k][n]=xb[n][k]
#pragma unroll
    for (int t = 0; t < 8; ++t)  b[t]     = (_Float16)pb[8 * h + t];
#pragma unroll
    for (int t = 0; t < 8; ++t)  b[8 + t] = (_Float16)pb[16 + 8 * h + t];

    v8f c = {};
    c = __builtin_amdgcn_wmma_f32_16x16x32_f16(false, a, false, b, (short)0, c, false, false);

    // D layout: VGPR p -> row = ti*16 + p + 8h, col = tj*16 + m
    int j = tj * 16 + m;
    float nbj = nb[j];
#pragma unroll
    for (int p = 0; p < 8; ++p) {
        int i = ti * 16 + p + 8 * h;
        float d = na[i] + nbj - 2.0f * c[p];
        d = fmaxf(d, 0.0f);
        float k = __expf(-d * (1.0f / EPS));
        Kout[(size_t)i * N + j] = (_Float16)k;
    }
}

// ---------------------------------------------------------------------------
__global__ void init_ones_kernel(float* __restrict__ u0, float* __restrict__ u1,
                                 float* __restrict__ u2) {
    int i = blockIdx.x * blockDim.x + threadIdx.x;
    if (i < N) { u0[i] = 1.0f; u1[i] = 1.0f; u2[i] = 1.0f; }
}

// ---------------------------------------------------------------------------
// out[r] = (1/N) / (sum_j K[r][j]*w[j] + STAB).  One wave per row; lanes read
// contiguous 512B chunks (32 lanes x 16B of fp16). Pure HBM stream.
__global__ void rowdot_kernel(const _Float16* __restrict__ K, const float* __restrict__ w,
                              float* __restrict__ out) {
    int wave = (blockIdx.x * blockDim.x + threadIdx.x) >> 5;   // row, 0..N-1
    int lane = threadIdx.x & 31;
    const h8* row = (const h8*)(K + (size_t)wave * N);
    const f4* wf  = (const f4*)w;
    float acc = 0.f;
#pragma unroll 4
    for (int it = 0; it < N / 256; ++it) {
        int idx = it * 32 + lane;
        h8 kv = row[idx];
        f4 w0 = wf[idx * 2];
        f4 w1 = wf[idx * 2 + 1];
        acc += (float)kv[0] * w0.x + (float)kv[1] * w0.y + (float)kv[2] * w0.z + (float)kv[3] * w0.w
             + (float)kv[4] * w1.x + (float)kv[5] * w1.y + (float)kv[6] * w1.z + (float)kv[7] * w1.w;
    }
#pragma unroll
    for (int off = 16; off >= 1; off >>= 1) acc += __shfl_xor(acc, off, 32);
    if (lane == 0) out[wave] = (1.0f / (float)N) / (acc + STABF);
}

// ---------------------------------------------------------------------------
// partial[block] = sum over its 8 rows r of u[r] * sum_j K*ln(K)*v[j]
// (cost recovered as -EPS*ln(K); the -EPS factor is applied at finalize).
__global__ void plancost_kernel(const _Float16* __restrict__ K, const float* __restrict__ u,
                                const float* __restrict__ v, float* __restrict__ partial) {
    int wave = (blockIdx.x * blockDim.x + threadIdx.x) >> 5;
    int lane = threadIdx.x & 31;
    int widx = threadIdx.x >> 5;
    const h8* row = (const h8*)(K + (size_t)wave * N);
    const f4* vf  = (const f4*)v;
    float acc = 0.f;
    for (int it = 0; it < N / 256; ++it) {
        int idx = it * 32 + lane;
        h8 kv = row[idx];
        f4 v0 = vf[idx * 2];
        f4 v1 = vf[idx * 2 + 1];
        float vv[8] = { v0.x, v0.y, v0.z, v0.w, v1.x, v1.y, v1.z, v1.w };
#pragma unroll
        for (int e = 0; e < 8; ++e) {
            float k = (float)kv[e];
            if (k > 0.f) acc += k * __logf(k) * vv[e];   // k->0 limit: term -> 0
        }
    }
#pragma unroll
    for (int off = 16; off >= 1; off >>= 1) acc += __shfl_xor(acc, off, 32);
    __shared__ float wsum[8];
    if (lane == 0) wsum[widx] = acc * u[wave];
    __syncthreads();
    if (threadIdx.x == 0) {
        float s = 0.f;
#pragma unroll
        for (int q = 0; q < 8; ++q) s += wsum[q];
        partial[blockIdx.x] = s;
    }
}

// ---------------------------------------------------------------------------
// Deterministic serial reduction + debiased combine.
__global__ void finalize_kernel(const float* __restrict__ p_st, const float* __restrict__ p_ss,
                                const float* __restrict__ p_tt, float* __restrict__ out) {
    if (threadIdx.x == 0 && blockIdx.x == 0) {
        float sst = 0.f, sss = 0.f, stt = 0.f;
        for (int i = 0; i < 1024; ++i) sst += p_st[i];
        for (int i = 0; i < 1024; ++i) sss += p_ss[i];
        for (int i = 0; i < 1024; ++i) stt += p_tt[i];
        float wst = -EPS * sst;
        float wss = -EPS * sss;
        float wtt = -EPS * stt;
        out[0] = (wst - 0.5f * wss - 0.5f * wtt) / (float)N;
    }
}

// ---------------------------------------------------------------------------
extern "C" void kernel_launch(void* const* d_in, const int* in_sizes, int n_in,
                              void* d_out, int out_size, void* d_ws, size_t ws_size,
                              hipStream_t stream) {
    (void)in_sizes; (void)n_in; (void)out_size; (void)ws_size;
    const float* src = (const float*)d_in[0];
    const float* tgt = (const float*)d_in[1];
    float* outp = (float*)d_out;

    char* ws = (char*)d_ws;
    const size_t S = (size_t)N * N * sizeof(_Float16);   // 134 MB per matrix
    _Float16* Kst = (_Float16*)(ws);
    _Float16* Kts = (_Float16*)(ws + S);
    _Float16* Kss = (_Float16*)(ws + 2 * S);
    _Float16* Ktt = (_Float16*)(ws + 3 * S);
    float* ns  = (float*)(ws + 4 * S);
    float* nt  = ns  + N;
    float* ust = nt  + N;  float* vst = ust + N;
    float* uss = vst + N;  float* vss = uss + N;
    float* utt = vss + N;  float* vtt = utt + N;
    float* pst = vtt + N;          // 1024 partials each
    float* pss = pst + 1024;
    float* ptt = pss + 1024;

    // Setup: norms + four fp16 kernel matrices (WMMA Gram + exp).
    norms_kernel<<<N / 256, 256, 0, stream>>>(src, ns);
    norms_kernel<<<N / 256, 256, 0, stream>>>(tgt, nt);
    const int gramBlocks = (512 * 512) / 8;              // 8 waves/block, 1 tile/wave
    gram_exp_kernel<<<gramBlocks, 256, 0, stream>>>(src, tgt, ns, nt, Kst);
    gram_exp_kernel<<<gramBlocks, 256, 0, stream>>>(tgt, src, nt, ns, Kts);  // = Kst^T
    gram_exp_kernel<<<gramBlocks, 256, 0, stream>>>(src, src, ns, ns, Kss);
    gram_exp_kernel<<<gramBlocks, 256, 0, stream>>>(tgt, tgt, nt, nt, Ktt);
    init_ones_kernel<<<N / 256, 256, 0, stream>>>(ust, uss, utt);

    // 100 Sinkhorn iterations per pair; symmetric K for ss/tt so K^T u == K u.
    for (int it = 0; it < NUM_ITERS; ++it) {
        rowdot_kernel<<<1024, 256, 0, stream>>>(Kts, ust, vst);  // v = nu/(K^T u)
        rowdot_kernel<<<1024, 256, 0, stream>>>(Kst, vst, ust);  // u = mu/(K v)
        rowdot_kernel<<<1024, 256, 0, stream>>>(Kss, uss, vss);
        rowdot_kernel<<<1024, 256, 0, stream>>>(Kss, vss, uss);
        rowdot_kernel<<<1024, 256, 0, stream>>>(Ktt, utt, vtt);
        rowdot_kernel<<<1024, 256, 0, stream>>>(Ktt, vtt, utt);
    }

    // sum(plan * cost) per pair, then debiased combine.
    plancost_kernel<<<1024, 256, 0, stream>>>(Kst, ust, vst, pst);
    plancost_kernel<<<1024, 256, 0, stream>>>(Kss, uss, vss, pss);
    plancost_kernel<<<1024, 256, 0, stream>>>(Ktt, utt, vtt, ptt);
    finalize_kernel<<<1, 64, 0, stream>>>(pst, pss, ptt, outp);
}
